// Graphormer_87273735454954
// MI455X (gfx1250) — matrix operand order repaired and checked by reading, not compiled, not run on hardware
//
#include <hip/hip_runtime.h>
#include <hip/hip_bf16.h>

typedef __attribute__((ext_vector_type(16))) _Float16 v16h;
typedef __attribute__((ext_vector_type(8)))  _Float16 half8;
typedef __attribute__((ext_vector_type(8)))  float    v8f;
typedef __attribute__((ext_vector_type(4)))  unsigned uint32x4;
typedef __attribute__((ext_vector_type(8)))  int      int32x8;
typedef __attribute__((ext_vector_type(4)))  int      int32x4;

#define HIDC 128
#define HEADS 8
#define CDIM 16

#if defined(__HIP_DEVICE_COMPILE__) && __has_builtin(__builtin_amdgcn_tensor_load_to_lds)
#define HAVE_TDM 1
#else
#define HAVE_TDM 0
#endif

// ---------- helpers ----------
static __device__ __forceinline__ unsigned ordEnc(float f) {
    unsigned u = __float_as_uint(f);
    return (u & 0x80000000u) ? ~u : (u | 0x80000000u);
}
static __device__ __forceinline__ float ordDec(unsigned u) {
    u = (u & 0x80000000u) ? (u & 0x7FFFFFFFu) : ~u;
    return __uint_as_float(u);
}

// ---------- tiny utility kernels ----------
__global__ void fill_u32_kernel(unsigned* __restrict__ p, unsigned v, long n) {
    long i = (long)blockIdx.x * blockDim.x + threadIdx.x;
    if (i < n) p[i] = v;
}

__global__ void cvt_f32_to_f16_kernel(const float* __restrict__ a, _Float16* __restrict__ b, long n) {
    long i = (long)blockIdx.x * blockDim.x + threadIdx.x;
    if (i < n) b[i] = (_Float16)a[i];
}

// transpose 128x128 f32 row-major -> f16 column-major (WT[n*128+k] = W[k*128+n])
__global__ void wtrans_kernel(const float* __restrict__ W, _Float16* __restrict__ WT, int n) {
    int i = blockIdx.x * blockDim.x + threadIdx.x;
    if (i >= n) return;
    int k = i >> 7, c = i & 127;
    WT[(size_t)c * HIDC + k] = (_Float16)W[i];
}

// ---------- degree centrality ----------
__global__ void deg_accum_kernel(const int* __restrict__ src, float* __restrict__ deg, long E) {
    long i = (long)blockIdx.x * blockDim.x + threadIdx.x;
    if (i < E) atomicAdd(&deg[src[i]], 1.0f);
}
__global__ void deg_max_kernel(const float* __restrict__ deg, unsigned* __restrict__ mb, long N) {
    long i = (long)blockIdx.x * blockDim.x + threadIdx.x;
    if (i < N) atomicMax(mb, __float_as_uint(deg[i]));   // deg >= 0: raw bits order-preserving
}
__global__ void deg_norm_kernel(float* __restrict__ deg, const unsigned* __restrict__ mb, long N) {
    long i = (long)blockIdx.x * blockDim.x + threadIdx.x;
    if (i >= N) return;
    float m = fmaxf(__uint_as_float(*mb), 1.0f);
    deg[i] = deg[i] / m;
}

// ---------- WMMA GEMM: C[M,128] = A_f16[M,128] @ WT_f16[128,128] (+bias +deg*lastrow) ----------
// Block = 256 threads = 8 waves. Block covers 128 rows: wave w owns rows [bid*128+16w, +16).
// B operand (whole 32KB WT) is staged into LDS once per block via the Tensor Data Mover
// (fallback: cooperative b128 copy), then each wave loops all 8 column tiles reusing its
// register-cached A fragments -> 32 v_wmma per wave, B reads are ds_load_b128.
__global__ void __launch_bounds__(256)
wmma_gemm128_kernel(const _Float16* __restrict__ A, const _Float16* __restrict__ WT,
                    const float* __restrict__ bias, const float* __restrict__ rowScale,
                    const float* __restrict__ colVec, float* __restrict__ Cf32,
                    _Float16* __restrict__ Cf16, int M, int relu)
{
    __shared__ _Float16 lds_wt[HIDC * HIDC];

#if HAVE_TDM
    if (threadIdx.x < 32) {
        // Tensor DMA descriptor (D#), groups 0/1 per CDNA5 ISA §8; 2D 128x128 x 2B tile.
        const unsigned lds_off = (unsigned)(unsigned long long)(const void*)lds_wt;
        const unsigned long long ga = (unsigned long long)WT;
        uint32x4 g0;
        g0[0] = 1u;                                                   // count=1, user D#
        g0[1] = lds_off;                                              // lds_addr
        g0[2] = (unsigned)ga;                                         // global_addr[31:0]
        g0[3] = (unsigned)((ga >> 32) & 0x01FFFFFFull) | 0x80000000u; // global_addr[56:32] | type=2
        int32x8 g1;
        g1[0] = 0x00010000;          // workgroup_mask=0, data_size=1 (2 bytes)
        g1[1] = (int)(128u << 16);   // tensor_dim0 = 128 (bits 79:48 -> low16 here)
        g1[2] = (int)(128u << 16);   // tensor_dim1 = 128 (bits 111:80)
        g1[3] = (int)(128u << 16);   // tile_dim0 = 128 (bits 127:112)
        g1[4] = 128;                 // tile_dim1 = 128 (bits 143:128)
        g1[5] = 128;                 // tensor_dim0_stride = 128 (bits 207:160)
        g1[6] = 0;
        g1[7] = 0;
        int32x4 z4 = {};
#if defined(__clang_major__) && (__clang_major__ >= 23)
        int32x8 z8 = {};
        __builtin_amdgcn_tensor_load_to_lds(g0, g1, z4, z4, z8, 0);
#else
        __builtin_amdgcn_tensor_load_to_lds(g0, g1, z4, z4, 0);
#endif
        __builtin_amdgcn_s_wait_tensorcnt(0);
    }
    __syncthreads();
#else
    {
        const half8* s8 = (const half8*)WT;
        half8*       d8 = (half8*)lds_wt;
        for (int i = threadIdx.x; i < (HIDC * HIDC) / 8; i += 256) d8[i] = s8[i];
    }
    __syncthreads();
#endif

    const int lane   = threadIdx.x & 31;
    const int wave   = threadIdx.x >> 5;
    const int m_base = blockIdx.x * 128 + wave * 16;
    const int g      = lane >> 4;       // half-wave group
    const int l15    = lane & 15;

    int arow = m_base + l15;
    if (arow >= M) arow = M - 1;        // clamp (tail block); stores are guarded below

    // Register-cache the wave's A fragments for all 4 K-steps (16x32 f16 layout, ISA §7.12.2):
    // lanes 0-15 hold K = k0+0..7 & k0+16..23; lanes 16-31 hold +8.
    const half8* Arow = (const half8*)(A + (size_t)arow * HIDC);
    half8 afr[8];
#pragma unroll
    for (int kk = 0; kk < 4; ++kk) {
        afr[2 * kk]     = Arow[kk * 4 + g];
        afr[2 * kk + 1] = Arow[kk * 4 + 2 + g];
    }

#pragma unroll
    for (int nt = 0; nt < 8; ++nt) {    // all 8 column tiles, A reused from registers
        const int n = nt * 16 + l15;
        const half8* Bcol = (const half8*)(lds_wt + (size_t)n * HIDC);
        v8f acc = {};
#pragma unroll
        for (int kk = 0; kk < 4; ++kk) {
            // B 32x16 fragment from LDS: lanes 0-15 K=k0+0..7 & k0+16..23; lanes 16-31 hold +8
            half8 blo = Bcol[kk * 4 + g];
            half8 bhi = Bcol[kk * 4 + 2 + g];
            v16h af, bf;
#pragma unroll
            for (int i = 0; i < 8; ++i) {
                af[i] = afr[2 * kk][i]; af[i + 8] = afr[2 * kk + 1][i];
                bf[i] = blo[i];         bf[i + 8] = bhi[i];
            }
            acc = __builtin_amdgcn_wmma_f32_16x16x32_f16(false, af, false, bf,
                                                         (short)0, acc, false, false);
        }
        const float bv = bias   ? bias[n]   : 0.0f;
        const float cv = colVec ? colVec[n] : 0.0f;
#pragma unroll
        for (int r = 0; r < 8; ++r) {
            const int rr = m_base + r + 8 * g;   // C/D layout: VGPR r -> M=r (+8 upper lanes)
            if (rr < M) {
                float val = acc[r] + bv;
                if (rowScale) val += rowScale[rr] * cv;
                if (relu) val = fmaxf(val, 0.0f);
                const size_t off = (size_t)rr * HIDC + n;
                if (Cf32) Cf32[off] = val;
                if (Cf16) Cf16[off] = (_Float16)val;
            }
        }
    }
}

// ---------- edge pass 1: alpha + segment max ----------
__global__ void edge_alpha_kernel(const int* __restrict__ src, const int* __restrict__ dst,
                                  const float* __restrict__ eattr,
                                  const _Float16* __restrict__ q, const _Float16* __restrict__ k,
                                  const float* __restrict__ werow,
                                  float* __restrict__ alpha, unsigned* __restrict__ mbits, long EH)
{
    long i = (long)blockIdx.x * blockDim.x + threadIdx.x;
    if (i >= EH) return;
    const long e  = i >> 3;
    const int  hd = (int)(i & 7);
    const int  s = src[e], d = dst[e];
    const float ea = eattr[e];
    const half8* qd  = (const half8*)(q + (size_t)d * HIDC + hd * CDIM);
    const half8* ksp = (const half8*)(k + (size_t)s * HIDC + hd * CDIM);
    const float* we  = werow + hd * CDIM;
    half8 q0 = qd[0], q1 = qd[1], k0 = ksp[0], k1 = ksp[1];
    float acc = 0.0f;
#pragma unroll
    for (int c = 0; c < 8; ++c) {
        acc += (float)q0[c] * ((float)k0[c] + ea * we[c]);
        acc += (float)q1[c] * ((float)k1[c] + ea * we[c + 8]);
    }
    acc *= 0.25f;                       // 1/sqrt(C), C = 16
    alpha[i] = acc;
    atomicMax(&mbits[(size_t)d * HEADS + hd], ordEnc(acc));
}

// ---------- edge pass 2: exp, denominator, unnormalized value scatter ----------
__global__ void edge_scatter_kernel(const int* __restrict__ src, const int* __restrict__ dst,
                                    const float* __restrict__ eattr,
                                    const _Float16* __restrict__ v, const float* __restrict__ werow,
                                    const float* __restrict__ alpha, const unsigned* __restrict__ mbits,
                                    float* __restrict__ den, float* __restrict__ num, long EH)
{
    long i = (long)blockIdx.x * blockDim.x + threadIdx.x;
    if (i >= EH) return;
    const long e  = i >> 3;
    const int  hd = (int)(i & 7);
    const int  s = src[e], d = dst[e];
    const float ea = eattr[e];
    const float m  = ordDec(mbits[(size_t)d * HEADS + hd]);
    const float ex = __expf(alpha[i] - m);
    atomicAdd(&den[(size_t)d * HEADS + hd], ex);
    const half8* vs = (const half8*)(v + (size_t)s * HIDC + hd * CDIM);
    const float* we = werow + hd * CDIM;
    half8 v0 = vs[0], v1 = vs[1];
    float* np = num + (size_t)d * HIDC + hd * CDIM;
#pragma unroll
    for (int c = 0; c < 8; ++c) {
        atomicAdd(&np[c],     ex * ((float)v0[c] + ea * we[c]));
        atomicAdd(&np[c + 8], ex * ((float)v1[c] + ea * we[c + 8]));
    }
}

// ---------- node combine: softmax normalize + skip + ReLU -> h (f16) ----------
__global__ void combine_kernel(const float* __restrict__ num, const float* __restrict__ den,
                               const float* __restrict__ skip, _Float16* __restrict__ h16, long NC)
{
    long i = (long)blockIdx.x * blockDim.x + threadIdx.x;
    if (i >= NC) return;
    long nnode = i >> 7;
    int  ch = (int)(i & 127);
    int  hd = ch >> 4;
    float dn  = den[nnode * HEADS + hd];
    float att = (dn > 0.0f) ? num[i] / dn : 0.0f;
    float val = att + skip[i];
    h16[i] = (_Float16)fmaxf(val, 0.0f);
}

// ---------- global add pool ----------
__global__ void pool_kernel(const _Float16* __restrict__ h16, const int* __restrict__ batch,
                            float* __restrict__ gbuf, long NC)
{
    long i = (long)blockIdx.x * blockDim.x + threadIdx.x;
    if (i >= NC) return;
    long n = i >> 7;
    int ch = (int)(i & 127);
    atomicAdd(&gbuf[(size_t)batch[n] * HIDC + ch], (float)h16[i]);
}

// ---------- classifier (tiny: G x 10) ----------
__global__ void cls_kernel(const _Float16* __restrict__ r, const float* __restrict__ Wc,
                           const float* __restrict__ bc, float* __restrict__ out, int G, int OUT)
{
    int i = blockIdx.x * blockDim.x + threadIdx.x;
    if (i >= G * OUT) return;
    int gph = i / OUT, o = i % OUT;
    float acc = bc[o];
    const _Float16* rr = r + (size_t)gph * HIDC;
#pragma unroll 4
    for (int k = 0; k < HIDC; ++k) acc += (float)rr[k] * Wc[k * OUT + o];
    out[i] = acc;
}

// ---------- host side ----------
static inline void* bump(char*& p, size_t bytes) {
    void* r = (void*)p;
    p += (bytes + 255) & ~(size_t)255;
    return r;
}
static inline unsigned gblk(long n) { return (unsigned)((n + 255) / 256); }
static inline unsigned gblk128(long m) { return (unsigned)((m + 127) / 128); }

extern "C" void kernel_launch(void* const* d_in, const int* in_sizes, int n_in,
                              void* d_out, int out_size, void* d_ws, size_t ws_size,
                              hipStream_t stream) {
    const float* x      = (const float*)d_in[0];
    const int*   ei     = (const int*)  d_in[1];
    const int*   batch  = (const int*)  d_in[2];
    const float* eattr  = (const float*)d_in[3];
    const float* W_in   = (const float*)d_in[4];
    const float* b_in   = (const float*)d_in[5];
    const float* Wq     = (const float*)d_in[6];
    const float* bq     = (const float*)d_in[7];
    const float* Wk     = (const float*)d_in[8];
    const float* bk     = (const float*)d_in[9];
    const float* Wv     = (const float*)d_in[10];
    const float* bv     = (const float*)d_in[11];
    const float* We     = (const float*)d_in[12];
    const float* Wskip  = (const float*)d_in[13];
    const float* bskip  = (const float*)d_in[14];
    const float* W_r1   = (const float*)d_in[15];
    const float* b_r1   = (const float*)d_in[16];
    const float* W_r2   = (const float*)d_in[17];
    const float* b_r2   = (const float*)d_in[18];
    const float* W_cls  = (const float*)d_in[19];
    const float* b_cls  = (const float*)d_in[20];
    float* out = (float*)d_out;

    const long N   = in_sizes[0] / HIDC;
    const long E   = in_sizes[1] / 2;
    const int  L   = in_sizes[6] / (HIDC * HIDC);
    const int  OUT = in_sizes[20];
    const int  G   = out_size / OUT;

    const int* src = ei;
    const int* dst = ei + E;
    const long NC = N * HIDC;
    const long EH = E * HEADS;

    // ---- workspace layout ----
    char* p = (char*)d_ws;
    _Float16* x16   = (_Float16*)bump(p, (size_t)NC * 2);
    _Float16* h16   = (_Float16*)bump(p, (size_t)NC * 2);
    _Float16* q16   = (_Float16*)bump(p, (size_t)NC * 2);
    _Float16* k16   = (_Float16*)bump(p, (size_t)NC * 2);
    _Float16* v16   = (_Float16*)bump(p, (size_t)NC * 2);
    float*    skipf = (float*)   bump(p, (size_t)NC * 4);
    float*    num   = (float*)   bump(p, (size_t)NC * 4);
    float*    alpha = (float*)   bump(p, (size_t)EH * 4);
    unsigned* mbits = (unsigned*)bump(p, (size_t)N * HEADS * 4);
    float*    den   = (float*)   bump(p, (size_t)N * HEADS * 4);
    float*    deg   = (float*)   bump(p, (size_t)N * 4);
    unsigned* dmax  = (unsigned*)bump(p, 4);
    float*    gbuf  = (float*)   bump(p, (size_t)G * HIDC * 4);
    _Float16* g16   = (_Float16*)bump(p, (size_t)G * HIDC * 2);
    _Float16* r1b   = (_Float16*)bump(p, (size_t)G * HIDC * 2);
    _Float16* r2b   = (_Float16*)bump(p, (size_t)G * HIDC * 2);
    _Float16* WTin  = (_Float16*)bump(p, (size_t)HIDC * HIDC * 2);
    _Float16* WTq   = (_Float16*)bump(p, (size_t)HIDC * HIDC * 2);
    _Float16* WTk   = (_Float16*)bump(p, (size_t)HIDC * HIDC * 2);
    _Float16* WTv   = (_Float16*)bump(p, (size_t)HIDC * HIDC * 2);
    _Float16* WTs   = (_Float16*)bump(p, (size_t)HIDC * HIDC * 2);
    _Float16* WTr1  = (_Float16*)bump(p, (size_t)HIDC * HIDC * 2);
    _Float16* WTr2  = (_Float16*)bump(p, (size_t)HIDC * HIDC * 2);
    (void)ws_size;

    const int WW = HIDC * HIDC;

    // ---- degree centrality ----
    fill_u32_kernel<<<gblk(N), 256, 0, stream>>>((unsigned*)deg, 0u, N);
    fill_u32_kernel<<<1, 256, 0, stream>>>(dmax, 0u, 1);
    deg_accum_kernel<<<gblk(E), 256, 0, stream>>>(src, deg, E);
    deg_max_kernel<<<gblk(N), 256, 0, stream>>>(deg, dmax, N);
    deg_norm_kernel<<<gblk(N), 256, 0, stream>>>(deg, dmax, N);

    // ---- input projection: h = [x | deg] @ W_in + b_in ----
    cvt_f32_to_f16_kernel<<<gblk(NC), 256, 0, stream>>>(x, x16, NC);
    wtrans_kernel<<<gblk(WW), 256, 0, stream>>>(W_in, WTin, WW);   // first 128 rows
    wmma_gemm128_kernel<<<gblk128(N), 256, 0, stream>>>(
        x16, WTin, b_in, deg, W_in + (size_t)HIDC * HIDC, nullptr, h16, (int)N, 0);

    // ---- transformer layers ----
    for (int l = 0; l < L; ++l) {
        const size_t wo = (size_t)l * WW;
        const size_t bo = (size_t)l * HIDC;
        wtrans_kernel<<<gblk(WW), 256, 0, stream>>>(Wq + wo, WTq, WW);
        wtrans_kernel<<<gblk(WW), 256, 0, stream>>>(Wk + wo, WTk, WW);
        wtrans_kernel<<<gblk(WW), 256, 0, stream>>>(Wv + wo, WTv, WW);
        wtrans_kernel<<<gblk(WW), 256, 0, stream>>>(Wskip + wo, WTs, WW);

        wmma_gemm128_kernel<<<gblk128(N), 256, 0, stream>>>(
            h16, WTq, bq + bo, nullptr, nullptr, nullptr, q16, (int)N, 0);
        wmma_gemm128_kernel<<<gblk128(N), 256, 0, stream>>>(
            h16, WTk, bk + bo, nullptr, nullptr, nullptr, k16, (int)N, 0);
        wmma_gemm128_kernel<<<gblk128(N), 256, 0, stream>>>(
            h16, WTv, bv + bo, nullptr, nullptr, nullptr, v16, (int)N, 0);
        wmma_gemm128_kernel<<<gblk128(N), 256, 0, stream>>>(
            h16, WTs, bskip + bo, nullptr, nullptr, skipf, nullptr, (int)N, 0);

        fill_u32_kernel<<<gblk(N * HEADS), 256, 0, stream>>>(mbits, 0x007FFFFFu, N * HEADS); // ordEnc(-inf)
        fill_u32_kernel<<<gblk(N * HEADS), 256, 0, stream>>>((unsigned*)den, 0u, N * HEADS);
        fill_u32_kernel<<<gblk(NC), 256, 0, stream>>>((unsigned*)num, 0u, NC);

        const float* werow = We + bo;   // We[l] is (1,128): broadcast edge embedding row
        edge_alpha_kernel<<<gblk(EH), 256, 0, stream>>>(src, dst, eattr, q16, k16, werow,
                                                        alpha, mbits, EH);
        edge_scatter_kernel<<<gblk(EH), 256, 0, stream>>>(src, dst, eattr, v16, werow,
                                                          alpha, mbits, den, num, EH);
        combine_kernel<<<gblk(NC), 256, 0, stream>>>(num, den, skipf, h16, NC);
    }

    // ---- readout ----
    fill_u32_kernel<<<gblk((long)G * HIDC), 256, 0, stream>>>((unsigned*)gbuf, 0u, (long)G * HIDC);
    pool_kernel<<<gblk(NC), 256, 0, stream>>>(h16, batch, gbuf, NC);
    cvt_f32_to_f16_kernel<<<gblk((long)G * HIDC), 256, 0, stream>>>(gbuf, g16, (long)G * HIDC);

    wtrans_kernel<<<gblk(WW), 256, 0, stream>>>(W_r1, WTr1, WW);
    wtrans_kernel<<<gblk(WW), 256, 0, stream>>>(W_r2, WTr2, WW);
    wmma_gemm128_kernel<<<gblk128(G), 256, 0, stream>>>(
        g16, WTr1, b_r1, nullptr, nullptr, nullptr, r1b, G, 1);
    wmma_gemm128_kernel<<<gblk128(G), 256, 0, stream>>>(
        r1b, WTr2, b_r2, nullptr, nullptr, nullptr, r2b, G, 1);

    cls_kernel<<<gblk((long)G * OUT), 256, 0, stream>>>(r2b, W_cls, b_cls, out, G, OUT);
}
